// RiemGrassAtt_987842478681
// MI455X (gfx1250) — matrix operand order repaired and checked by reading, not compiled, not run on hardware
//
#include <hip/hip_runtime.h>
#include <math.h>

// ---------------------------------------------------------------- types
typedef __attribute__((ext_vector_type(16))) __bf16 v16bf;
typedef __attribute__((ext_vector_type(8)))  float  v8f;
typedef __attribute__((ext_vector_type(4)))  float  f32x4;
typedef __attribute__((ext_vector_type(4)))  unsigned int u32x4;
typedef __attribute__((ext_vector_type(4)))  int    i32x4;
typedef __attribute__((ext_vector_type(8)))  int    i32x8;

// Constants for this problem (B=4, N=1024, C=512, H=8, d=64)
#define BATCH 4
#define SEQ   1024
#define CDIM  512
#define NH    8
#define HD    64
#define NCH   16   // 2*NH

static __device__ __forceinline__ unsigned short f2bf(float f) {
  unsigned int u = __float_as_uint(f);
  unsigned int r = (u + 0x7FFFu + ((u >> 16) & 1u)) >> 16;
  return (unsigned short)r;
}
// pack two f32 -> one dword of two bf16 (lo = first)
static __device__ __forceinline__ unsigned int pk_bf2(float lo, float hi) {
  return (unsigned int)f2bf(lo) | ((unsigned int)f2bf(hi) << 16);
}

static __device__ __forceinline__ v8f wmma_bf(v16bf a, v16bf b, v8f c) {
  return __builtin_amdgcn_wmma_f32_16x16x32_bf16(
      /*neg_a=*/false, a, /*neg_b=*/false, b,
      /*c_mod=*/(short)0, c, /*reuse_a=*/false, /*reuse_b=*/false);
}

// ----------------- fragment loaders (ISA 7.12.2 VGPR layouts, wave32) -----
// A 16x32 (16-bit): lane r=lane&15 holds row M=r; lanes<16 K={kb..kb+7, kb+16..kb+23}
// with kb=0, lanes>=16 kb=8.  Two contiguous 16B runs -> two b128 loads.
static __device__ __forceinline__ v16bf load_a_bf(const unsigned short* p, int ld, int lane) {
  union { v16bf v; u32x4 q[2]; } f;
  int r = lane & 15; int kb = (lane >> 4) ? 8 : 0;
  const unsigned short* row = p + (long)r * ld + kb;
  f.q[0] = *(const u32x4*)(row);        // K kb..kb+7
  f.q[1] = *(const u32x4*)(row + 16);   // K kb+16..kb+23
  return f.v;
}
static __device__ __forceinline__ v16bf load_a_f32(const float* p, int ld, int lane) {
  union { v16bf v; unsigned int w[8]; } f;
  int r = lane & 15; int kb = (lane >> 4) ? 8 : 0;
  const float* row = p + (long)r * ld + kb;
  f32x4 a0 = *(const f32x4*)(row);
  f32x4 a1 = *(const f32x4*)(row + 4);
  f32x4 a2 = *(const f32x4*)(row + 16);
  f32x4 a3 = *(const f32x4*)(row + 20);
  f.w[0] = pk_bf2(a0[0], a0[1]); f.w[1] = pk_bf2(a0[2], a0[3]);
  f.w[2] = pk_bf2(a1[0], a1[1]); f.w[3] = pk_bf2(a1[2], a1[3]);
  f.w[4] = pk_bf2(a2[0], a2[1]); f.w[5] = pk_bf2(a2[2], a2[3]);
  f.w[6] = pk_bf2(a3[0], a3[1]); f.w[7] = pk_bf2(a3[2], a3[3]);
  return f.v;
}
// A-transposed source (strided): element (r,k) = p[k*ld + r]  (for Gram Q^T Q)
static __device__ __forceinline__ v16bf load_aT_bf(const unsigned short* p, int ld, int lane) {
  union { v16bf v; unsigned short u[16]; } f;
  int r = lane & 15; int kb = (lane >> 4) ? 8 : 0;
#pragma unroll
  for (int v = 0; v < 8; ++v) {
    int k = ((v & 3) << 1) + ((v >> 2) << 4) + kb;
    f.u[2 * v]     = p[(long)k * ld + r];
    f.u[2 * v + 1] = p[(long)(k + 1) * ld + r];
  }
  return f.v;
}
// B 32x16 (16-bit): lane holds col N=lane&15; K={kb..kb+7, kb+16..kb+23}, kb=0 (lanes<16) or 8.
// (strided column read; kept scalar)
static __device__ __forceinline__ v16bf load_b_bf(const unsigned short* p, int ld, int lane) {
  union { v16bf v; unsigned short u[16]; } f;
  int c = lane & 15; int kb = (lane >> 4) << 3;
#pragma unroll
  for (int h = 0; h < 16; ++h) f.u[h] = p[(long)(kb + h + ((h >> 3) << 3)) * ld + c];
  return f.v;
}
// B from transposed (row-major [N,K]) source: element (k,c) = p[c*ld + k]
// -> two contiguous 16B runs per lane (K kb..kb+7, kb+16..kb+23): two b128 loads.
static __device__ __forceinline__ v16bf load_bT_bf(const unsigned short* p, int ld, int lane) {
  union { v16bf v; u32x4 q[2]; } f;
  int c = lane & 15; int kb = (lane >> 4) << 3;
  const unsigned short* row = p + (long)c * ld + kb;
  f.q[0] = *(const u32x4*)(row);
  f.q[1] = *(const u32x4*)(row + 16);
  return f.v;
}
static __device__ __forceinline__ v16bf load_bT_f32(const float* p, int ld, int lane) {
  union { v16bf v; unsigned int w[8]; } f;
  int c = lane & 15; int kb = (lane >> 4) << 3;
  const float* row = p + (long)c * ld + kb;
  f32x4 a0 = *(const f32x4*)(row);
  f32x4 a1 = *(const f32x4*)(row + 4);
  f32x4 a2 = *(const f32x4*)(row + 16);
  f32x4 a3 = *(const f32x4*)(row + 20);
  f.w[0] = pk_bf2(a0[0], a0[1]); f.w[1] = pk_bf2(a0[2], a0[3]);
  f.w[2] = pk_bf2(a1[0], a1[1]); f.w[3] = pk_bf2(a1[2], a1[3]);
  f.w[4] = pk_bf2(a2[0], a2[1]); f.w[5] = pk_bf2(a2[2], a2[3]);
  f.w[6] = pk_bf2(a3[0], a3[1]); f.w[7] = pk_bf2(a3[2], a3[3]);
  return f.v;
}
// C/D 16x16 f32: VGPR g -> M = g + (lane>=16 ? 8 : 0), N = lane&15
static __device__ __forceinline__ void store_c(float* p, int ld, int lane, v8f c) {
  int n = lane & 15; int mb = (lane >> 4) << 3;
#pragma unroll
  for (int g = 0; g < 8; ++g) p[(long)(mb + g) * ld + n] = c[g];
}
static __device__ __forceinline__ void store_c_bf(unsigned short* p, int ld, int lane, v8f c) {
  int n = lane & 15; int mb = (lane >> 4) << 3;
#pragma unroll
  for (int g = 0; g < 8; ++g) p[(long)(mb + g) * ld + n] = f2bf(c[g]);
}

// ---------------------------------------------------------------- kernels
__global__ void k_zero(float* p, int n) {
  int i = blockIdx.x * blockDim.x + threadIdx.x;
  if (i < n) p[i] = 0.f;
}

// K1: qkv = x @ W^T + b ; scatter to q,k (f32+bf16) and v (bf16) in [B,H,N,d]
__global__ void k_qkv(const float* __restrict__ x, const float* __restrict__ w,
                      const float* __restrict__ bias,
                      float* __restrict__ qf, float* __restrict__ kf,
                      unsigned short* __restrict__ qb, unsigned short* __restrict__ kb,
                      unsigned short* __restrict__ vb) {
  int lane = threadIdx.x & 31, wave = threadIdx.x >> 5;
  long t = (long)blockIdx.x * 8 + wave;          // 256*96 tiles
  int tm = (int)(t / 96), tn = (int)(t % 96);
  int m0 = tm * 16, n0 = tn * 16;
  v8f acc = {};
  for (int k0 = 0; k0 < CDIM; k0 += 32) {
    if (k0 + 32 < CDIM) {
      __builtin_prefetch((const void*)(x + (long)m0 * CDIM + k0 + 32), 0, 1);
      __builtin_prefetch((const void*)(w + (long)n0 * CDIM + k0 + 32), 0, 1);
    }
    v16bf a = load_a_f32(x + (long)m0 * CDIM + k0, CDIM, lane);
    v16bf b = load_bT_f32(w + (long)n0 * CDIM + k0, CDIM, lane);
    acc = wmma_bf(a, b, acc);
  }
  int n = lane & 15, mb = (lane >> 4) << 3;
  int j = n0 + n;
  float bj = bias[j];
  int which = j >> 9, h = (j >> 6) & 7, dd = j & 63;
#pragma unroll
  for (int g = 0; g < 8; ++g) {
    int m = m0 + mb + g;                // 0..4095
    int b_ = m >> 10, ns = m & 1023;
    float val = acc[g] + bj;
    long idx = (((long)(b_ * NH + h)) * SEQ + ns) * HD + dd;
    unsigned short bf = f2bf(val);
    if (which == 0)      { qf[idx] = val; qb[idx] = bf; }
    else if (which == 1) { kf[idx] = val; kb[idx] = bf; }
    else                 { vb[idx] = bf; }
  }
}

// K2: center rows over d, store bf16 (packed dword stores)
__global__ void k_center(const float* __restrict__ qf, const float* __restrict__ kf,
                         unsigned short* __restrict__ qc, unsigned short* __restrict__ kc) {
  long r = (long)blockIdx.x * blockDim.x + threadIdx.x;   // 2 * 32768 rows
  const float* src = (r < 32768) ? qf : kf;
  unsigned short* dst = (r < 32768) ? qc : kc;
  long row = r & 32767;
  const float* p = src + row * HD;
  float s = 0.f;
#pragma unroll
  for (int i = 0; i < HD; i += 4) {
    f32x4 v4 = *(const f32x4*)(p + i);
    s += v4[0] + v4[1] + v4[2] + v4[3];
  }
  s *= (1.0f / (float)HD);
  unsigned int* o = (unsigned int*)(dst + row * HD);
#pragma unroll
  for (int i = 0; i < HD; i += 4) {
    f32x4 v4 = *(const f32x4*)(p + i);
    o[i / 2]     = pk_bf2(v4[0] - s, v4[1] - s);
    o[i / 2 + 1] = pk_bf2(v4[2] - s, v4[3] - s);
  }
}

// K3: Gram G = Q^T Q per (side,b,h) [64x64], K=1024
__global__ void k_gram(const unsigned short* __restrict__ qb,
                       const unsigned short* __restrict__ kb, float* __restrict__ G) {
  int lane = threadIdx.x & 31, wave = threadIdx.x >> 5;
  long t = (long)blockIdx.x * 8 + wave;        // 64 mats * 16 tiles
  int mat = (int)(t >> 4); int tt = (int)(t & 15);
  int ti = tt >> 2, tj = tt & 3;
  const unsigned short* Q = ((mat < 32) ? qb : kb) + (long)(mat & 31) * SEQ * HD;
  v8f acc = {};
  for (int k0 = 0; k0 < SEQ; k0 += 32) {
    v16bf a = load_aT_bf(Q + (long)k0 * HD + ti * 16, HD, lane);
    v16bf b = load_b_bf (Q + (long)k0 * HD + tj * 16, HD, lane);
    acc = wmma_bf(a, b, acc);
  }
  store_c(G + (long)mat * 4096 + ti * 16 * HD + tj * 16, HD, lane, acc);
}

// K4: Cholesky (upper R: R^T R = G) + invert R -> U (bf16).
// G tile is DMA'd into LDS with the Tensor Data Mover (TDM) when available.
__global__ void k_cholinv(const float* __restrict__ G, unsigned short* __restrict__ U) {
  __shared__ __align__(16) float A[64 * 64];
  __shared__ __align__(16) float V[64 * 64];
  int mat = blockIdx.x, tid = threadIdx.x;
  const float* g = G + (long)mat * 4096;
#if __has_builtin(__builtin_amdgcn_tensor_load_to_lds) && __has_builtin(__builtin_amdgcn_s_wait_tensorcnt)
  if (tid < 32) {   // wave-uniform: only wave 0 issues the TDM op (EXEC is ignored by TDM)
    // D# group 0: count=1 | lds_addr | global_addr(57b) | type=2
    unsigned int lds_off = (unsigned int)(unsigned long long)&A[0];
    unsigned long long ga = (unsigned long long)g;
    u32x4 d0;
    d0[0] = 1u;                                   // count=1, is_restore=0, gather off
    d0[1] = lds_off;                              // bits 63:32  lds_addr
    d0[2] = (unsigned int)(ga & 0xffffffffu);     // bits 95:64  global_addr lo
    d0[3] = (unsigned int)((ga >> 32) & 0x1ffffffu) | (2u << 30);  // addr hi | type=2
    // D# group 1: data_size=4B(code2); tensor 64x64, tile 64x64, stride0=64
    i32x8 d1;
    d1[0] = (int)(2u << 16);       // wg_mask=0 | data_size=2
    d1[1] = (int)(64u << 16);      // tensor_dim0[15:0] in hi16
    d1[2] = (int)(64u << 16);      // tensor_dim1[15:0] in hi16
    d1[3] = (int)(64u << 16);      // tile_dim0 in hi16
    d1[4] = 64;                    // tile_dim1 lo16, tile_dim2=0
    d1[5] = 64;                    // tensor_dim0_stride lo32
    d1[6] = 0;
    d1[7] = 0;
    i32x4 z4 = {0, 0, 0, 0};
#if defined(__clang_major__) && (__clang_major__ >= 23)
    i32x8 z8 = {0, 0, 0, 0, 0, 0, 0, 0};
    __builtin_amdgcn_tensor_load_to_lds(d0, d1, z4, z4, z8, 0);
#else
    __builtin_amdgcn_tensor_load_to_lds(d0, d1, z4, z4, 0);
#endif
    __builtin_amdgcn_s_wait_tensorcnt(0);
  }
  __syncthreads();
#else
  for (int i = tid; i < 4096; i += 64) A[i] = g[i];
  __syncthreads();
#endif
  for (int kc = 0; kc < 64; ++kc) {
    if (tid == 0) A[kc * 64 + kc] = sqrtf(fmaxf(A[kc * 64 + kc], 1e-12f));
    __syncthreads();
    float inv = 1.0f / A[kc * 64 + kc];
    if (tid > kc) A[kc * 64 + tid] *= inv;
    __syncthreads();
    if (tid > kc) {
      float akt = A[kc * 64 + tid];
      for (int r2 = kc + 1; r2 <= tid; ++r2)
        A[r2 * 64 + tid] -= A[kc * 64 + r2] * akt;
    }
    __syncthreads();
  }
  { // invert upper-triangular, column tid
    int j = tid;
    for (int i = 0; i < 64; ++i) V[i * 64 + j] = 0.f;
    V[j * 64 + j] = 1.0f / A[j * 64 + j];
    for (int i = j - 1; i >= 0; --i) {
      float s = 0.f;
      for (int t2 = i + 1; t2 <= j; ++t2) s += A[i * 64 + t2] * V[t2 * 64 + j];
      V[i * 64 + j] = -s / A[i * 64 + i];
    }
  }
  __syncthreads();
  unsigned short* u = U + (long)mat * 4096;
  for (int i = tid; i < 4096; i += 64) u[i] = f2bf(V[i]);
}

// K5: Qgr = Q @ R^-1  ([1024x64]x[64x64]) per matrix
__global__ void k_applyR(const unsigned short* __restrict__ qb,
                         const unsigned short* __restrict__ kb,
                         const unsigned short* __restrict__ U,
                         unsigned short* __restrict__ qgr, unsigned short* __restrict__ kgr) {
  int lane = threadIdx.x & 31, wave = threadIdx.x >> 5;
  long t = (long)blockIdx.x * 8 + wave;        // 64 mats * 256 tiles
  int mat = (int)(t >> 8); int tt = (int)(t & 255);
  int ti = tt >> 2, tj = tt & 3;
  const unsigned short* Q = ((mat < 32) ? qb : kb) + (long)(mat & 31) * SEQ * HD;
  unsigned short* O = ((mat < 32) ? qgr : kgr) + (long)(mat & 31) * SEQ * HD;
  const unsigned short* u = U + (long)mat * 4096;
  v8f acc = {};
  for (int k0 = 0; k0 < HD; k0 += 32) {
    v16bf a = load_a_bf(Q + (long)ti * 16 * HD + k0, HD, lane);
    v16bf b = load_b_bf(u + (long)k0 * HD + tj * 16, HD, lane);
    acc = wmma_bf(a, b, acc);
  }
  store_c_bf(O + (long)ti * 16 * HD + tj * 16, HD, lane, acc);
}

// K6: per-channel BN statistics over recomputed score tiles
__global__ void k_stats(const unsigned short* __restrict__ qc, const unsigned short* __restrict__ kc,
                        const unsigned short* __restrict__ qgr, const unsigned short* __restrict__ kgr,
                        const float* __restrict__ rs, const float* __restrict__ gs,
                        float* __restrict__ stats) {
  int lane = threadIdx.x & 31, wave = threadIdx.x >> 5;
  long t = (long)blockIdx.x * 8 + wave;        // B*16*64*64
  int tm = (int)(t & 63); long r1 = t >> 6;
  int tn = (int)(r1 & 63); r1 >>= 6;
  int c = (int)(r1 & 15); int bb = (int)(r1 >> 4);
  bool riem = c < 8; int h = c & 7;
  long base = ((long)(bb * NH + h)) * SEQ * HD;
  const unsigned short* Am = riem ? qc + base : qgr + base;
  const unsigned short* Bm = riem ? kc + base : kgr + base;
  v8f acc = {};
  for (int k0 = 0; k0 < HD; k0 += 32) {
    v16bf a = load_a_bf (Am + (long)tn * 16 * HD + k0, HD, lane);
    v16bf b = load_bT_bf(Bm + (long)tm * 16 * HD + k0, HD, lane);
    acc = wmma_bf(a, b, acc);
  }
  float scale = riem ? (rs[0] * (1.0f / (float)(HD - 1))) : gs[0];
  float s = 0.f, s2 = 0.f;
#pragma unroll
  for (int g = 0; g < 8; ++g) {
    float vv = acc[g];
    float a_ = riem ? vv * scale : vv * vv * scale;
    s += a_; s2 += a_ * a_;
  }
  for (int off = 16; off; off >>= 1) {
    s  += __shfl_xor(s,  off, 32);
    s2 += __shfl_xor(s2, off, 32);
  }
  if (lane == 0) { atomicAdd(&stats[c * 2], s); atomicAdd(&stats[c * 2 + 1], s2); }
}

// K7: fold BN affine + 1x1 conv -> Wp[8][16], bp[8]
__global__ void k_bnfold(const float* __restrict__ stats, const float* __restrict__ gamma,
                         const float* __restrict__ beta, const float* __restrict__ cw,
                         const float* __restrict__ cb, float* __restrict__ Wp,
                         float* __restrict__ bp) {
  __shared__ float s_[NCH], t_[NCH];
  int tid = threadIdx.x;
  if (tid < NCH) {
    float cnt = (float)BATCH * (float)SEQ * (float)SEQ;
    float mu  = stats[tid * 2] / cnt;
    float var = stats[tid * 2 + 1] / cnt - mu * mu;
    float sc  = gamma[tid] * rsqrtf(var + 1e-5f);
    s_[tid] = sc; t_[tid] = beta[tid] - mu * sc;
  }
  __syncthreads();
  if (tid < NH * NCH) Wp[tid] = cw[tid] * s_[tid & 15];
  if (tid < NH) {
    float acc = cb[tid];
    for (int c = 0; c < NCH; ++c) acc += cw[tid * NCH + c] * t_[c];
    bp[tid] = acc;
  }
}

// K8: logits[b,h] tile = sum_c Wp[h][c]*a_c + bp[h]; 16 channel tiles per wave
__global__ void k_logits(const unsigned short* __restrict__ qc, const unsigned short* __restrict__ kc,
                         const unsigned short* __restrict__ qgr, const unsigned short* __restrict__ kgr,
                         const float* __restrict__ rs, const float* __restrict__ gs,
                         const float* __restrict__ Wp, const float* __restrict__ bp,
                         float* __restrict__ logits) {
  int lane = threadIdx.x & 31, wave = threadIdx.x >> 5;
  long t = (long)blockIdx.x * 8 + wave;        // B*64*64
  int tm = (int)(t & 63); long r1 = t >> 6;
  int tn = (int)(r1 & 63); int bb = (int)(r1 >> 6);
  float rscale = rs[0] * (1.0f / (float)(HD - 1)), gscale = gs[0];
  v8f zero = {};
  v8f hacc[NH];
#pragma unroll
  for (int h = 0; h < NH; ++h) hacc[h] = zero;
  for (int c = 0; c < NCH; ++c) {
    bool riem = c < 8; int hh = c & 7;
    long base = ((long)(bb * NH + hh)) * SEQ * HD;
    const unsigned short* Am = riem ? qc + base : qgr + base;
    const unsigned short* Bm = riem ? kc + base : kgr + base;
    if (c + 1 < NCH) {   // prefetch next channel's panels
      bool r2 = (c + 1) < 8; int h2 = (c + 1) & 7;
      long b2 = ((long)(bb * NH + h2)) * SEQ * HD;
      __builtin_prefetch((const void*)((r2 ? qc : qgr) + b2 + (long)tn * 16 * HD), 0, 1);
      __builtin_prefetch((const void*)((r2 ? kc : kgr) + b2 + (long)tm * 16 * HD), 0, 1);
    }
    v8f acc = zero;
    for (int k0 = 0; k0 < HD; k0 += 32) {
      v16bf a = load_a_bf (Am + (long)tn * 16 * HD + k0, HD, lane);
      v16bf b = load_bT_bf(Bm + (long)tm * 16 * HD + k0, HD, lane);
      acc = wmma_bf(a, b, acc);
    }
#pragma unroll
    for (int g = 0; g < 8; ++g) {
      float vv = acc[g];
      acc[g] = riem ? vv * rscale : vv * vv * gscale;
    }
#pragma unroll
    for (int h = 0; h < NH; ++h) {
      float w = Wp[h * NCH + c];
#pragma unroll
      for (int g = 0; g < 8; ++g) hacc[h][g] += w * acc[g];
    }
  }
#pragma unroll
  for (int h = 0; h < NH; ++h) {
    float bh = bp[h];
    v8f o;
#pragma unroll
    for (int g = 0; g < 8; ++g) o[g] = hacc[h][g] + bh;
    store_c(logits + (((long)(bb * NH + h)) * SEQ + (long)tn * 16) * SEQ + tm * 16,
            SEQ, lane, o);
  }
}

// K9: softmax over m (unnormalized exp in LDS bf16) + AV via WMMA; fold 1/sum in epilogue
__global__ void k_softmax_av(const float* __restrict__ logits,
                             const unsigned short* __restrict__ vb,
                             unsigned short* __restrict__ ao) {
  __shared__ __align__(16) unsigned short P[16 * SEQ];   // 32KB exp(l - max) as bf16
  __shared__ float red[16 * 8];
  __shared__ float invs[16];
  int tid = threadIdx.x;                      // 128 threads = 4 waves
  int bb = blockIdx.x >> 9;
  int h  = (blockIdx.x >> 6) & 7;
  int rt = blockIdx.x & 63;
  const float* L = logits + (((long)(bb * NH + h)) * SEQ + (long)rt * 16) * SEQ;
  int row = tid >> 3, sub = tid & 7;          // 8 threads per row
  float mx = -3.0e38f;
  for (int m = sub * 4; m < SEQ; m += 32) {   // vectorized row scan
    f32x4 v4 = *(const f32x4*)(L + (long)row * SEQ + m);
    mx = fmaxf(mx, fmaxf(fmaxf(v4[0], v4[1]), fmaxf(v4[2], v4[3])));
  }
  red[row * 8 + sub] = mx;
  __syncthreads();
  if (sub == 0) {
    float m2 = red[row * 8];
    for (int i = 1; i < 8; ++i) m2 = fmaxf(m2, red[row * 8 + i]);
    red[row * 8] = m2;
  }
  __syncthreads();
  mx = red[row * 8];
  __syncthreads();
  float sum = 0.f;
  for (int m = sub * 4; m < SEQ; m += 32) {
    f32x4 v4 = *(const f32x4*)(L + (long)row * SEQ + m);
    float e0 = __expf(v4[0] - mx), e1 = __expf(v4[1] - mx);
    float e2 = __expf(v4[2] - mx), e3 = __expf(v4[3] - mx);
    unsigned int* pp = (unsigned int*)&P[row * SEQ + m];
    pp[0] = pk_bf2(e0, e1);
    pp[1] = pk_bf2(e2, e3);
    sum += e0 + e1 + e2 + e3;
  }
  red[row * 8 + sub] = sum;
  __syncthreads();
  if (sub == 0) {
    float s2 = 0.f;
    for (int i = 0; i < 8; ++i) s2 += red[row * 8 + i];
    invs[row] = 1.0f / s2;
  }
  __syncthreads();
  // AV: wave j owns d-tile j (16 cols), 16 rows, K=1024
  int lane = tid & 31, wave = tid >> 5;
  const unsigned short* V = vb + ((long)(bb * NH + h)) * SEQ * HD;
  v8f acc = {};
  for (int k0 = 0; k0 < SEQ; k0 += 32) {
    v16bf a = load_a_bf((const unsigned short*)P + k0, SEQ, lane);
    v16bf b = load_b_bf(V + (long)k0 * HD + wave * 16, HD, lane);
    acc = wmma_bf(a, b, acc);
  }
  int n_ = lane & 15, mb = (lane >> 4) << 3;
#pragma unroll
  for (int g = 0; g < 8; ++g) {
    int r2 = mb + g;
    float val = acc[g] * invs[r2];
    long n = (long)rt * 16 + r2;
    ao[((long)bb * SEQ + n) * CDIM + h * HD + wave * 16 + n_] = f2bf(val);
  }
}

// K10: y = ao @ proj_w^T + proj_b  (f32 out)
__global__ void k_proj(const unsigned short* __restrict__ ao, const float* __restrict__ w,
                       const float* __restrict__ bias, float* __restrict__ out) {
  int lane = threadIdx.x & 31, wave = threadIdx.x >> 5;
  long t = (long)blockIdx.x * 8 + wave;        // 256*32 tiles
  int tm = (int)(t >> 5), tn = (int)(t & 31);
  int m0 = tm * 16, n0 = tn * 16;
  v8f acc = {};
  for (int k0 = 0; k0 < CDIM; k0 += 32) {
    v16bf a = load_a_bf (ao + (long)m0 * CDIM + k0, CDIM, lane);
    v16bf b = load_bT_f32(w + (long)n0 * CDIM + k0, CDIM, lane);
    acc = wmma_bf(a, b, acc);
  }
  int n_ = lane & 15, mb = (lane >> 4) << 3;
  float bj = bias[n0 + n_];
#pragma unroll
  for (int g = 0; g < 8; ++g)
    out[(long)(m0 + mb + g) * CDIM + n0 + n_] = acc[g] + bj;
}

// ---------------------------------------------------------------- launch
extern "C" void kernel_launch(void* const* d_in, const int* in_sizes, int n_in,
                              void* d_out, int out_size, void* d_ws, size_t ws_size,
                              hipStream_t stream) {
  (void)in_sizes; (void)n_in; (void)out_size; (void)ws_size;
  const float* x      = (const float*)d_in[0];
  const float* qkv_w  = (const float*)d_in[1];
  const float* qkv_b  = (const float*)d_in[2];
  const float* rs     = (const float*)d_in[3];
  const float* gs     = (const float*)d_in[4];
  const float* gamma  = (const float*)d_in[5];
  const float* beta   = (const float*)d_in[6];
  const float* conv_w = (const float*)d_in[7];
  const float* conv_b = (const float*)d_in[8];
  const float* proj_w = (const float*)d_in[9];
  const float* proj_b = (const float*)d_in[10];
  float* out = (float*)d_out;

  const long NE = (long)BATCH * NH * SEQ * HD;   // 2M elements per [B,H,N,d] tensor
  char* p = (char*)d_ws;
  float*          qf     = (float*)p;          p += NE * 4;
  float*          kf     = (float*)p;          p += NE * 4;
  unsigned short* qb     = (unsigned short*)p; p += NE * 2;
  unsigned short* kb     = (unsigned short*)p; p += NE * 2;
  unsigned short* vb     = (unsigned short*)p; p += NE * 2;
  unsigned short* qc     = (unsigned short*)p; p += NE * 2;
  unsigned short* kc     = (unsigned short*)p; p += NE * 2;
  unsigned short* qgr    = (unsigned short*)p; p += NE * 2;
  unsigned short* kgr    = (unsigned short*)p; p += NE * 2;
  float*          G      = (float*)p;          p += 64L * 4096 * 4;
  unsigned short* U      = (unsigned short*)p; p += 64L * 4096 * 2;
  float*          stats  = (float*)p;          p += 256;
  float*          Wp     = (float*)p;          p += 128 * 4;
  float*          bp     = (float*)p;          p += 256;
  unsigned short* ao     = (unsigned short*)p; p += NE * 2;
  float*          logits = (float*)p;          // 32M floats (128MB)

  k_zero<<<1, 64, 0, stream>>>(stats, 32);
  k_qkv<<<3072, 256, 0, stream>>>(x, qkv_w, qkv_b, qf, kf, qb, kb, vb);
  k_center<<<256, 256, 0, stream>>>(qf, kf, qc, kc);
  k_gram<<<128, 256, 0, stream>>>(qb, kb, G);
  k_cholinv<<<64, 64, 0, stream>>>(G, U);
  k_applyR<<<2048, 256, 0, stream>>>(qb, kb, U, qgr, kgr);
  k_stats<<<32768, 256, 0, stream>>>(qc, kc, qgr, kgr, rs, gs, stats);
  k_bnfold<<<1, 128, 0, stream>>>(stats, gamma, beta, conv_w, conv_b, Wp, bp);
  k_logits<<<2048, 256, 0, stream>>>(qc, kc, qgr, kgr, rs, gs, Wp, bp, logits);
  k_softmax_av<<<2048, 128, 0, stream>>>(logits, vb, ao);
  k_proj<<<1024, 256, 0, stream>>>(ao, proj_w, proj_b, out);
}
